// GNN_auto_38439957299727
// MI455X (gfx1250) — matrix operand order repaired and checked by reading, not compiled, not run on hardware
//
#include <hip/hip_runtime.h>
#include <hip/hip_bf16.h>

typedef __attribute__((ext_vector_type(16))) __bf16 v16bf;
typedef __attribute__((ext_vector_type(8)))  float  v8f;

__device__ __forceinline__ __bf16 f2bf(float f) {
    // native convert: lowers to v_cvt_*bf16_f32 (RNE) on gfx1250
    return (__bf16)f;
}

__device__ __forceinline__ v8f splat8(float x) {
    v8f r = {x, x, x, x, x, x, x, x};
    return r;
}

__device__ __forceinline__ v8f wmma_bf16(v16bf a, v16bf b, v8f c) {
    return __builtin_amdgcn_wmma_f32_16x16x32_bf16(false, a, false, b, (short)0, c,
                                                   false, false);
}

__device__ __forceinline__ float sigmoidf_(float x) { return 1.0f / (1.0f + __expf(-x)); }

// ---------------------------------------------------------------------------
// zero buffer
__global__ void zero_kernel(float* p, long n) {
    long i = (long)blockIdx.x * blockDim.x + threadIdx.x;
    if (i < n) p[i] = 0.0f;
}

// hidden[qsub[q]] = qre[q_rel[q]]
__global__ void init_hidden_kernel(float* hidden, const float* qre, const int* q_rel,
                                   const int* qsub, int nq) {
    int idx = blockIdx.x * blockDim.x + threadIdx.x;
    if (idx >= nq * 64) return;
    int q = idx >> 6, d = idx & 63;
    hidden[(long)qsub[q] * 64 + d] = qre[(long)q_rel[q] * 64 + d];
}

// ---------------------------------------------------------------------------
// Pack W (outdim x 64, row-major) into bf16 B-operand layout for
// v_wmma_f32_16x16x32_bf16.  outB index: (((t*2+kc)*32 + lane)*16 + h)
// B is 32x16 (KxN): col N=lane%16; K=(h&7)+16*(h>>3)+8*(lane>>4).
__global__ void prep_B_kernel(const float* __restrict__ W, __bf16* __restrict__ outB,
                              int ntiles) {
    int idx = blockIdx.x * blockDim.x + threadIdx.x;
    int total = ntiles * 2 * 32;
    if (idx >= total) return;
    int lane = idx & 31;
    int kc = (idx >> 5) & 1;
    int t = idx >> 6;
    int half = lane >> 4, n = lane & 15;
#pragma unroll
    for (int h = 0; h < 16; ++h) {
        int K = kc * 32 + (h & 7) + ((h >> 3) << 4) + (half << 3);
        outB[(long)idx * 16 + h] = f2bf(W[(long)(t * 16 + n) * 64 + K]);
    }
}

// ---------------------------------------------------------------------------
// Edge phase: one wave per edge (2 dims per lane), attention + sigmoid gate,
// scatter-add alpha*(hs*hr) into agg via f32 atomics.
__global__ __launch_bounds__(256) void edge_kernel(
    const float* __restrict__ hidden, const float* __restrict__ re,
    const float* __restrict__ Ws, const float* __restrict__ Wr,
    const float* __restrict__ Wqr, const float* __restrict__ Wqrb,
    const float* __restrict__ waW, const float* __restrict__ wab,
    const int* __restrict__ edges, const int* __restrict__ q_rel,
    const int* __restrict__ ebi, float* __restrict__ agg, int nE) {
    __shared__ float sWs[320], sWr[320], sWq[320], sb[5], swa[5], swab[1];
    int t = threadIdx.x;
    for (int i = t; i < 320; i += 256) {
        sWs[i] = Ws[i];
        sWr[i] = Wr[i];
        sWq[i] = Wqr[i];
    }
    if (t < 5) {
        sb[t] = Wqrb[t];
        swa[t] = waW[t];
    }
    if (t == 0) swab[0] = wab[0];
    __syncthreads();

    int lane = t & 31;
    int e = blockIdx.x * 8 + (t >> 5);
    if (e >= nE) return;

    int sub = edges[(long)e * 3 + 0];
    int rel = edges[(long)e * 3 + 1];
    int obj = edges[(long)e * 3 + 2];
    int qr = q_rel[ebi[e]];

    float hs0 = hidden[(long)sub * 64 + lane];
    float hs1 = hidden[(long)sub * 64 + 32 + lane];
    float hr0 = re[(long)rel * 64 + lane];
    float hr1 = re[(long)rel * 64 + 32 + lane];
    float hq0 = re[(long)qr * 64 + lane];
    float hq1 = re[(long)qr * 64 + 32 + lane];

    float att[5];
#pragma unroll
    for (int a = 0; a < 5; ++a) {
        float p = hs0 * sWs[a * 64 + lane] + hs1 * sWs[a * 64 + 32 + lane] +
                  hr0 * sWr[a * 64 + lane] + hr1 * sWr[a * 64 + 32 + lane] +
                  hq0 * sWq[a * 64 + lane] + hq1 * sWq[a * 64 + 32 + lane];
#pragma unroll
        for (int off = 16; off > 0; off >>= 1) p += __shfl_xor(p, off, 32);
        float av = p + sb[a];
        att[a] = av > 0.0f ? av : 0.0f;
    }
    float z = swab[0];
#pragma unroll
    for (int a = 0; a < 5; ++a) z += att[a] * swa[a];
    float alpha = sigmoidf_(z);

    atomicAdd(&agg[(long)obj * 64 + lane], alpha * hs0 * hr0);
    atomicAdd(&agg[(long)obj * 64 + 32 + lane], alpha * hs1 * hr1);
}

// ---------------------------------------------------------------------------
// Node phase: one wave per 16-node tile.
//  h = relu(agg @ Wh^T)           (4 col-tiles x 2 k-chunks WMMA)
//  GRU: rg/zg = sigmoid(h@Wih^T + h0@Whh^T + b), ng = tanh(i_n + rg*h_n)
//  hidden = h0 = ((1-zg)*ng + zg*h0) * mask
__global__ __launch_bounds__(32) void node_gru_kernel(
    const float* __restrict__ agg, const float* __restrict__ h0,
    float* __restrict__ hidden, float* __restrict__ h0out,
    const __bf16* __restrict__ WhB, const __bf16* __restrict__ WihB,
    const __bf16* __restrict__ WhhB, const float* __restrict__ bih,
    const float* __restrict__ bhh, int n_node) {
    __shared__ float lbuf[16 * 64];
    __shared__ float lmask[16];

    int lane = threadIdx.x & 31;
    int half = lane >> 4;
    int m = lane & 15;
    int row_base = blockIdx.x * 16;
    if (row_base >= n_node) return;
    bool full = (row_base + 16) <= n_node;

    // ---- stage agg tile into LDS (coalesced) ----
    if (full) {
        const float4* gv = (const float4*)(agg + (long)row_base * 64);
        float4* lv = (float4*)lbuf;
#pragma unroll
        for (int it = 0; it < 8; ++it) lv[it * 32 + lane] = gv[it * 32 + lane];
    } else {
        for (int it = 0; it < 32; ++it) {
            int idx = it * 32 + lane;
            int r = row_base + (idx >> 6);
            r = r < n_node ? r : (n_node - 1);
            lbuf[idx] = agg[(long)r * 64 + (idx & 63)];
        }
    }
    __syncthreads();

    // ---- build A-layout regs for agg ----
    v16bf aA[2];
#pragma unroll
    for (int kc = 0; kc < 2; ++kc)
#pragma unroll
        for (int h = 0; h < 16; ++h) {
            int kl = (h & 7) + ((h >> 3) << 4) + (half << 3);
            aA[kc][h] = f2bf(lbuf[m * 64 + kc * 32 + kl]);
        }
    __syncthreads();

    // ---- h = relu(agg @ Wh^T), write into LDS in row-major ----
#pragma unroll
    for (int nt = 0; nt < 4; ++nt) {
        v8f c = {};
#pragma unroll
        for (int kc = 0; kc < 2; ++kc) {
            v16bf b = *(const v16bf*)(WhB + (long)((nt * 2 + kc) * 32 + lane) * 16);
            c = wmma_bf16(aA[kc], b, c);
        }
#pragma unroll
        for (int v = 0; v < 8; ++v) {
            float x = c[v];
            x = x > 0.0f ? x : 0.0f;
            lbuf[(v + half * 8) * 64 + nt * 16 + m] = x;
        }
    }
    __syncthreads();

    // ---- per-node mask ----
    if (lane < 16) {
        float s = 0.0f;
        for (int d = 0; d < 64; ++d) s += lbuf[lane * 64 + d];
        lmask[lane] = (s != 0.0f) ? 1.0f : 0.0f;
    }
    __syncthreads();

    // ---- A-layout regs for h ----
    v16bf hA[2];
#pragma unroll
    for (int kc = 0; kc < 2; ++kc)
#pragma unroll
        for (int h = 0; h < 16; ++h) {
            int kl = (h & 7) + ((h >> 3) << 4) + (half << 3);
            hA[kc][h] = f2bf(lbuf[m * 64 + kc * 32 + kl]);
        }
    __syncthreads();

    // ---- stage h0 tile ----
    if (full) {
        const float4* gv = (const float4*)(h0 + (long)row_base * 64);
        float4* lv = (float4*)lbuf;
#pragma unroll
        for (int it = 0; it < 8; ++it) lv[it * 32 + lane] = gv[it * 32 + lane];
    } else {
        for (int it = 0; it < 32; ++it) {
            int idx = it * 32 + lane;
            int r = row_base + (idx >> 6);
            r = r < n_node ? r : (n_node - 1);
            lbuf[idx] = h0[(long)r * 64 + (idx & 63)];
        }
    }
    __syncthreads();

    // ---- A-layout + C-layout regs for h0 ----
    v16bf h0A[2];
#pragma unroll
    for (int kc = 0; kc < 2; ++kc)
#pragma unroll
        for (int h = 0; h < 16; ++h) {
            int kl = (h & 7) + ((h >> 3) << 4) + (half << 3);
            h0A[kc][h] = f2bf(lbuf[m * 64 + kc * 32 + kl]);
        }
    v8f h0c[4];
#pragma unroll
    for (int nt = 0; nt < 4; ++nt)
#pragma unroll
        for (int v = 0; v < 8; ++v) h0c[nt][v] = lbuf[(v + half * 8) * 64 + nt * 16 + m];

    // ---- GRU, one 16-column tile at a time ----
#pragma unroll
    for (int nt = 0; nt < 4; ++nt) {
        int colb = nt * 16 + m;
        // rg = sigmoid(h@Wih_r^T + h0@Whh_r^T + bih_r + bhh_r)
        v8f cr = splat8(bih[colb] + bhh[colb]);
#pragma unroll
        for (int kc = 0; kc < 2; ++kc) {
            v16bf bw = *(const v16bf*)(WihB + (long)(((0 * 4 + nt) * 2 + kc) * 32 + lane) * 16);
            cr = wmma_bf16(hA[kc], bw, cr);
            v16bf bh = *(const v16bf*)(WhhB + (long)(((0 * 4 + nt) * 2 + kc) * 32 + lane) * 16);
            cr = wmma_bf16(h0A[kc], bh, cr);
        }
#pragma unroll
        for (int v = 0; v < 8; ++v) cr[v] = sigmoidf_(cr[v]);

        // i_n, h_n
        v8f cin = splat8(bih[128 + colb]);
        v8f chn = splat8(bhh[128 + colb]);
#pragma unroll
        for (int kc = 0; kc < 2; ++kc) {
            v16bf bw = *(const v16bf*)(WihB + (long)(((2 * 4 + nt) * 2 + kc) * 32 + lane) * 16);
            cin = wmma_bf16(hA[kc], bw, cin);
            v16bf bh = *(const v16bf*)(WhhB + (long)(((2 * 4 + nt) * 2 + kc) * 32 + lane) * 16);
            chn = wmma_bf16(h0A[kc], bh, chn);
        }
        v8f ng;
#pragma unroll
        for (int v = 0; v < 8; ++v) ng[v] = tanhf(cin[v] + cr[v] * chn[v]);

        // zg
        v8f cz = splat8(bih[64 + colb] + bhh[64 + colb]);
#pragma unroll
        for (int kc = 0; kc < 2; ++kc) {
            v16bf bw = *(const v16bf*)(WihB + (long)(((1 * 4 + nt) * 2 + kc) * 32 + lane) * 16);
            cz = wmma_bf16(hA[kc], bw, cz);
            v16bf bh = *(const v16bf*)(WhhB + (long)(((1 * 4 + nt) * 2 + kc) * 32 + lane) * 16);
            cz = wmma_bf16(h0A[kc], bh, cz);
        }
#pragma unroll
        for (int v = 0; v < 8; ++v) {
            float z = sigmoidf_(cz[v]);
            float hn = (1.0f - z) * ng[v] + z * h0c[nt][v];
            float out = hn * lmask[v + half * 8];
            int r = row_base + v + half * 8;
            if (r < n_node) {
                hidden[(long)r * 64 + colb] = out;
                h0out[(long)r * 64 + colb] = out;
            }
        }
    }
}

// ---------------------------------------------------------------------------
// scores = hidden @ Wf^T ; scatter into out[batch, abs]
__global__ __launch_bounds__(256) void final_kernel(
    const float* __restrict__ hidden, const float* __restrict__ Wf,
    const int* __restrict__ batch_idxs, const int* __restrict__ abs_idxs,
    const int* __restrict__ n_ent_p, float* __restrict__ out, int n_node) {
    int lane = threadIdx.x & 31;
    int v = blockIdx.x * 8 + (threadIdx.x >> 5);
    if (v >= n_node) return;
    float p = hidden[(long)v * 64 + lane] * Wf[lane] +
              hidden[(long)v * 64 + 32 + lane] * Wf[32 + lane];
#pragma unroll
    for (int off = 16; off > 0; off >>= 1) p += __shfl_xor(p, off, 32);
    if (lane == 0) out[(long)batch_idxs[v] * n_ent_p[0] + abs_idxs[v]] = p;
}

// ---------------------------------------------------------------------------
extern "C" void kernel_launch(void* const* d_in, const int* in_sizes, int n_in,
                              void* d_out, int out_size, void* d_ws, size_t ws_size,
                              hipStream_t stream) {
    const float* rela = (const float*)d_in[0];
    const float* Ws = (const float*)d_in[1];
    const float* Wr = (const float*)d_in[2];
    const float* WqrW = (const float*)d_in[3];
    const float* Wqrb = (const float*)d_in[4];
    const float* waW = (const float*)d_in[5];
    const float* wab = (const float*)d_in[6];
    const float* Wh = (const float*)d_in[7];
    const float* gWih = (const float*)d_in[8];
    const float* gWhh = (const float*)d_in[9];
    const float* gbih = (const float*)d_in[10];
    const float* gbhh = (const float*)d_in[11];
    const float* qre = (const float*)d_in[12];
    const float* Wf = (const float*)d_in[13];
    const int* edges = (const int*)d_in[14];
    const int* q_rel = (const int*)d_in[15];
    const int* batch_idxs = (const int*)d_in[16];
    const int* abs_idxs = (const int*)d_in[17];
    const int* qsub = (const int*)d_in[18];
    const int* ebi = (const int*)d_in[19];
    const int* n_ent_p = (const int*)d_in[20];

    const int D = 64;
    int nE = in_sizes[14] / 3;
    int NN = in_sizes[16];
    int NQ_ = in_sizes[15];
    int Rdim = in_sizes[12] / D;
    int Lnum = in_sizes[0] / (Rdim * D);

    float* hidden = (float*)d_ws;
    float* h0 = hidden + (long)NN * D;
    float* agg = h0 + (long)NN * D;
    __bf16* WhB = (__bf16*)(agg + (long)NN * D);   // L * 4*2*32*16
    __bf16* WihB = WhB + (long)Lnum * 4096;        // 12*2*32*16
    __bf16* WhhB = WihB + 12288;

    long nd = (long)NN * D;
    zero_kernel<<<(int)((nd + 255) / 256), 256, 0, stream>>>(hidden, nd);
    zero_kernel<<<(int)((nd + 255) / 256), 256, 0, stream>>>(h0, nd);
    zero_kernel<<<(out_size + 255) / 256, 256, 0, stream>>>((float*)d_out, out_size);
    init_hidden_kernel<<<(NQ_ * 64 + 255) / 256, 256, 0, stream>>>(hidden, qre, q_rel,
                                                                   qsub, NQ_);
    // pack weights into WMMA B-operand layout (bf16)
    for (int l = 0; l < Lnum; ++l)
        prep_B_kernel<<<1, 256, 0, stream>>>(Wh + (long)l * D * D, WhB + (long)l * 4096, 4);
    prep_B_kernel<<<3, 256, 0, stream>>>(gWih, WihB, 12);
    prep_B_kernel<<<3, 256, 0, stream>>>(gWhh, WhhB, 12);

    int edge_blocks = (nE + 7) / 8;
    int node_blocks = (NN + 15) / 16;
    for (int l = 0; l < Lnum; ++l) {
        zero_kernel<<<(int)((nd + 255) / 256), 256, 0, stream>>>(agg, nd);
        edge_kernel<<<edge_blocks, 256, 0, stream>>>(
            hidden, rela + (long)l * Rdim * D, Ws + (long)l * 5 * D,
            Wr + (long)l * 5 * D, WqrW + (long)l * 5 * D, Wqrb + (long)l * 5,
            waW + (long)l * 5, wab + l, edges, q_rel, ebi, agg, nE);
        node_gru_kernel<<<node_blocks, 32, 0, stream>>>(
            agg, h0, hidden, h0, WhB + (long)l * 4096, WihB, WhhB, gbih, gbhh, NN);
    }

    final_kernel<<<(NN + 7) / 8, 256, 0, stream>>>(hidden, Wf, batch_idxs, abs_idxs,
                                                   n_ent_p, (float*)d_out, NN);
}
